// PointNetSetAbstraction_11450382811198
// MI455X (gfx1250) — matrix operand order repaired and checked by reading, not compiled, not run on hardware
//
#include <hip/hip_runtime.h>
#include <hip/hip_bf16.h>

// ---------------- types ----------------
typedef _Float16 v16h __attribute__((ext_vector_type(16)));
typedef _Float16 v8h  __attribute__((ext_vector_type(8)));
typedef float    v8f  __attribute__((ext_vector_type(8)));

// ---------------- problem constants ----------------
#define BB   16
#define NN   4096
#define SS   1024
#define KK   32                 // nsample
#define CC   64                 // input point feature channels
#define MM   (BB*SS*KK)         // 524288 rows of the big feature matrix
#define KPAD1 96                // 67 padded to 3*32
#define R2   0.01f
#define EPSF 1e-5f

// =======================================================================
// 0) zero the BN stat accumulators (must run every launch; graph-replayed)
// =======================================================================
__global__ void zero_stats_kernel(float* p, int n) {
    int i = blockIdx.x * blockDim.x + threadIdx.x;
    if (i < n) p[i] = 0.0f;
}

// =======================================================================
// 1) Farthest-point sampling: one block per batch, coords in registers,
//    block-wide argmax via LDS tree reduction. Writes new_xyz directly
//    into d_out (B,3,S) and center coords to ws for later kernels.
// =======================================================================
__global__ __launch_bounds__(256)
void fps_kernel(const float* __restrict__ xyz, float* __restrict__ nxyz_ws,
                float* __restrict__ out_newxyz) {
    __shared__ float redv[256];
    __shared__ int   redi[256];
    __shared__ float lp[3];
    const int b = blockIdx.x;
    const int t = threadIdx.x;
    const float* base = xyz + (size_t)b * 3 * NN;

    float px[16], py[16], pz[16], dd[16];
#pragma unroll
    for (int i = 0; i < 16; ++i) {
        int n = t + i * 256;
        px[i] = base[n];
        py[i] = base[NN + n];
        pz[i] = base[2 * NN + n];
        dd[i] = 1e10f;
    }
    // selection 0 is point index 0
    if (t == 0) {
        lp[0] = base[0]; lp[1] = base[NN]; lp[2] = base[2 * NN];
        out_newxyz[(size_t)b * 3 * SS + 0 * SS + 0] = base[0];
        out_newxyz[(size_t)b * 3 * SS + 1 * SS + 0] = base[NN];
        out_newxyz[(size_t)b * 3 * SS + 2 * SS + 0] = base[2 * NN];
        nxyz_ws[((size_t)b * SS + 0) * 3 + 0] = base[0];
        nxyz_ws[((size_t)b * SS + 0) * 3 + 1] = base[NN];
        nxyz_ws[((size_t)b * SS + 0) * 3 + 2] = base[2 * NN];
    }
    __syncthreads();

    for (int it = 1; it < SS; ++it) {
        const float lx = lp[0], ly = lp[1], lz = lp[2];
        float bv = -1.0f; int bn = 0;
#pragma unroll
        for (int i = 0; i < 16; ++i) {
            float dx = px[i] - lx, dy = py[i] - ly, dz = pz[i] - lz;
            float d = dx * dx + dy * dy + dz * dz;
            float o = dd[i];
            d = d < o ? d : o;
            dd[i] = d;
            int n = t + i * 256;
            if (d > bv) { bv = d; bn = n; }
        }
        redv[t] = bv; redi[t] = bn;
        __syncthreads();
        for (int off = 128; off > 0; off >>= 1) {
            if (t < off) {
                float v2 = redv[t + off]; int i2 = redi[t + off];
                if (v2 > redv[t] || (v2 == redv[t] && i2 < redi[t])) {
                    redv[t] = v2; redi[t] = i2;
                }
            }
            __syncthreads();
        }
        int last = redi[0];
        if (t == 0) {
            float cx = base[last], cy = base[NN + last], cz = base[2 * NN + last];
            lp[0] = cx; lp[1] = cy; lp[2] = cz;
            out_newxyz[(size_t)b * 3 * SS + 0 * SS + it] = cx;
            out_newxyz[(size_t)b * 3 * SS + 1 * SS + it] = cy;
            out_newxyz[(size_t)b * 3 * SS + 2 * SS + it] = cz;
            nxyz_ws[((size_t)b * SS + it) * 3 + 0] = cx;
            nxyz_ws[((size_t)b * SS + it) * 3 + 1] = cy;
            nxyz_ws[((size_t)b * SS + it) * 3 + 2] = cz;
        }
        __syncthreads();
    }
}

// =======================================================================
// 2) Ball query: one wave32 per (b,s). Scan N in index order; wave ballot
//    compaction reproduces "sorted candidates truncated at 32" semantics.
// =======================================================================
__global__ __launch_bounds__(256)
void ballq_kernel(const float* __restrict__ xyz, const float* __restrict__ nxyz_ws,
                  int* __restrict__ idxbuf) {
    int gwarp = (blockIdx.x * blockDim.x + threadIdx.x) >> 5;
    int lane  = threadIdx.x & 31;
    if (gwarp >= BB * SS) return;
    int b = gwarp >> 10;
    const float* base = xyz + (size_t)b * 3 * NN;
    float cx = nxyz_ws[(size_t)gwarp * 3 + 0];
    float cy = nxyz_ws[(size_t)gwarp * 3 + 1];
    float cz = nxyz_ws[(size_t)gwarp * 3 + 2];
    int* out = idxbuf + (size_t)gwarp * KK;

    int cnt = 0;
    int firstn = NN - 1;   // reference fallback when nothing is in range
    for (int n0 = 0; n0 < NN; n0 += 32) {
        int n = n0 + lane;
        float dx = base[n] - cx, dy = base[NN + n] - cy, dz = base[2 * NN + n] - cz;
        bool pred = (dx * dx + dy * dy + dz * dz) <= R2;
        unsigned mask = (unsigned)__ballot(pred);
        if (mask) {
            if (cnt == 0) firstn = n0 + (__ffs(mask) - 1);
            int pos  = __popc(mask & ((1u << lane) - 1u));
            int slot = cnt + pos;
            if (pred && slot < KK) out[slot] = n;
            cnt += __popc(mask);
            if (cnt >= KK) break;
        }
    }
    if (lane >= cnt && lane < KK) out[lane] = firstn;
}

// =======================================================================
// 3) Convert weights to f16, pad W1's K dim 67 -> 96
// =======================================================================
__global__ void prep_weights_kernel(const float* __restrict__ W1,
                                    const float* __restrict__ W2,
                                    const float* __restrict__ W3,
                                    _Float16* __restrict__ W1h,
                                    _Float16* __restrict__ W2h,
                                    _Float16* __restrict__ W3h) {
    int t = blockIdx.x * blockDim.x + threadIdx.x;
    const int n1 = 64 * KPAD1, n2 = 64 * 64, n3 = 128 * 64;
    if (t < n1) {
        int n = t / KPAD1, k = t % KPAD1;
        W1h[t] = (_Float16)(k < 67 ? W1[n * 67 + k] : 0.0f);
    } else if (t < n1 + n2) {
        int u = t - n1;  W2h[u] = (_Float16)W2[u];
    } else if (t < n1 + n2 + n3) {
        int u = t - n1 - n2;  W3h[u] = (_Float16)W3[u];
    }
}

// =======================================================================
// 4) Gather + center-subtract + concat -> X0 f16 [M, 96] (cols 67..95 = 0)
// =======================================================================
__global__ __launch_bounds__(256)
void gather_kernel(const float* __restrict__ xyz, const float* __restrict__ points,
                   const int* __restrict__ idxbuf, const float* __restrict__ nxyz_ws,
                   _Float16* __restrict__ X0) {
    size_t t = (size_t)blockIdx.x * blockDim.x + threadIdx.x;
    if (t >= (size_t)MM * 12) return;
    int m  = (int)(t / 12);
    int c0 = (int)(t % 12) * 8;
    int b = m >> 15;          // / (SS*KK)
    int s = (m >> 5) & (SS - 1);
    int id = idxbuf[m];
    const float* ctr = nxyz_ws + ((size_t)b * SS + s) * 3;

    v8h v;
#pragma unroll
    for (int j = 0; j < 8; ++j) {
        int col = c0 + j;
        float f;
        if (col < 3) {
            f = xyz[(size_t)b * 3 * NN + (size_t)col * NN + id] - ctr[col];
        } else if (col < 67) {
            int c = col - 3;
            f = points[(size_t)b * CC * NN + (size_t)c * NN + id];
        } else {
            f = 0.0f;
        }
        v[j] = (_Float16)f;
    }
    *(v8h*)(X0 + (size_t)m * KPAD1 + c0) = v;
}

// =======================================================================
// 5) WMMA GEMM: Y[M,NOUT] = X[M,KPAD] * Wh[NOUT,KPAD]^T  (f16 in, f32 acc)
//    plus per-channel sum / sum-of-squares accumulation for BN.
//    Block = 8 waves, each wave owns a 16-row strip; K via 16x16x32 WMMA.
// =======================================================================
template <int KPAD, int NOUT>
__global__ __launch_bounds__(256)
void gemm_wmma_kernel(const _Float16* __restrict__ X, const _Float16* __restrict__ Wh,
                      _Float16* __restrict__ Y, float* __restrict__ gSum,
                      float* __restrict__ gSq) {
    __shared__ float sSum[NOUT];
    __shared__ float sSq[NOUT];
    const int t = threadIdx.x;
    if (t < NOUT) { sSum[t] = 0.0f; sSq[t] = 0.0f; }
    __syncthreads();

    const int lane = t & 31;
    const int wv   = t >> 5;
    const int m0   = blockIdx.x * 128 + wv * 16;
    const int col  = lane & 15;
    const int kh   = lane >> 4;           // which half of the wave
    constexpr int KSTEPS = KPAD / 32;

    // A fragments: lane<16 holds row m, K {0..15}; lane>=16 holds {16..31}
    const _Float16* pa = X + (size_t)(m0 + col) * KPAD;
    v16h afrag[KSTEPS];
#pragma unroll
    for (int kk = 0; kk < KSTEPS; ++kk) {
        v8h lo = *(const v8h*)(pa + kk * 32 + kh * 16);
        v8h hi = *(const v8h*)(pa + kk * 32 + kh * 16 + 8);
        afrag[kk] = __builtin_shufflevector(lo, hi, 0, 1, 2, 3, 4, 5, 6, 7,
                                            8, 9, 10, 11, 12, 13, 14, 15);
    }

#pragma unroll
    for (int nt = 0; nt < NOUT / 16; ++nt) {
        v8f acc = {0.f, 0.f, 0.f, 0.f, 0.f, 0.f, 0.f, 0.f};
        const int n = nt * 16 + col;
        // B: lane holds column n, 16 contiguous K of W row n (B[k][n] = W[n][k])
        const _Float16* pb = Wh + (size_t)n * KPAD + kh * 16;
#pragma unroll
        for (int kk = 0; kk < KSTEPS; ++kk) {
            v16h bfrag = *(const v16h*)(pb + kk * 32);
            acc = __builtin_amdgcn_wmma_f32_16x16x32_f16(
                false, afrag[kk], false, bfrag, (short)0, acc, false, false);
        }
        // epilogue: store f16 + accumulate BN stats
        float s = 0.0f, sq = 0.0f;
#pragma unroll
        for (int v = 0; v < 8; ++v) {
            float f = acc[v];
            s += f; sq += f * f;
            int mrow = m0 + v + kh * 8;
            Y[(size_t)mrow * NOUT + n] = (_Float16)f;
        }
        atomicAdd(&sSum[n], s);       // ds_add_f32
        atomicAdd(&sSq[n], sq);
    }
    __syncthreads();
    if (t < NOUT) {
        atomicAdd(&gSum[t], sSum[t]); // one global atomic per channel per WG
        atomicAdd(&gSq[t],  sSq[t]);
    }
}

// =======================================================================
// 6) BN finalize: scale/shift per channel
// =======================================================================
__global__ void bn_finalize_kernel(const float* __restrict__ gSum, const float* __restrict__ gSq,
                                   const float* __restrict__ gamma, const float* __restrict__ beta,
                                   float* __restrict__ scale, float* __restrict__ shift,
                                   int nout, float invM) {
    int c = blockIdx.x * blockDim.x + threadIdx.x;
    if (c >= nout) return;
    float mu  = gSum[c] * invM;
    float var = gSq[c] * invM - mu * mu;
    float sc  = gamma[c] * rsqrtf(var + EPSF);
    scale[c] = sc;
    shift[c] = beta[c] - mu * sc;
}

// =======================================================================
// 7) In-place BN + ReLU on f16 activations
// =======================================================================
__global__ __launch_bounds__(256)
void bn_relu_kernel(_Float16* __restrict__ Y, const float* __restrict__ scale,
                    const float* __restrict__ shift, int nout, unsigned total8) {
    unsigned i = blockIdx.x * blockDim.x + threadIdx.x;
    if (i >= total8) return;
    size_t base = (size_t)i * 8;
    int c0 = (int)(base % (unsigned)nout);
    v8h v = *(v8h*)(Y + base);
#pragma unroll
    for (int j = 0; j < 8; ++j) {
        float f = (float)v[j] * scale[c0 + j] + shift[c0 + j];
        v[j] = (_Float16)(f > 0.0f ? f : 0.0f);
    }
    *(v8h*)(Y + base) = v;
}

// =======================================================================
// 8) Fused BN + ReLU + max-over-nsample + transposed store of new_points
//    d_out layout: [ new_xyz (B*3*S) | new_points (B*128*S) ]
// =======================================================================
__global__ __launch_bounds__(256)
void maxpool_kernel(const _Float16* __restrict__ Y3, const float* __restrict__ scale,
                    const float* __restrict__ shift, float* __restrict__ out) {
    int t = blockIdx.x * blockDim.x + threadIdx.x;
    if (t >= BB * SS * 128) return;
    int o  = t & 127;
    int bs = t >> 7;
    int b  = bs >> 10;
    int s  = bs & (SS - 1);
    float sc = scale[o], sh = shift[o];
    const _Float16* p = Y3 + (size_t)bs * KK * 128 + o;
    float m = 0.0f;                      // relu floor
#pragma unroll 4
    for (int k = 0; k < KK; ++k) {
        float f = (float)p[(size_t)k * 128] * sc + sh;
        if (f > m) m = f;
    }
    out[(size_t)BB * 3 * SS + ((size_t)b * 128 + o) * SS + s] = m;
}

// =======================================================================
// host launch
// =======================================================================
extern "C" void kernel_launch(void* const* d_in, const int* in_sizes, int n_in,
                              void* d_out, int out_size, void* d_ws, size_t ws_size,
                              hipStream_t stream) {
    const float* xyz    = (const float*)d_in[0];
    const float* points = (const float*)d_in[1];
    const float* W1 = (const float*)d_in[2];
    const float* g1 = (const float*)d_in[3];
    const float* b1 = (const float*)d_in[4];
    const float* W2 = (const float*)d_in[5];
    const float* g2 = (const float*)d_in[6];
    const float* b2 = (const float*)d_in[7];
    const float* W3 = (const float*)d_in[8];
    const float* g3 = (const float*)d_in[9];
    const float* b3 = (const float*)d_in[10];
    float* out = (float*)d_out;

    // ---- workspace layout (256B aligned) ----
    unsigned char* ws = (unsigned char*)d_ws;
    size_t off = 0;
    auto alloc = [&](size_t bytes) -> void* {
        void* p = ws + off;
        off += (bytes + 255) & ~(size_t)255;
        return p;
    };
    float*    nxyz   = (float*)alloc((size_t)BB * SS * 3 * sizeof(float));
    int*      idxbuf = (int*)alloc((size_t)MM * sizeof(int));
    _Float16* W1h    = (_Float16*)alloc((size_t)64 * KPAD1 * sizeof(_Float16));
    _Float16* W2h    = (_Float16*)alloc((size_t)64 * 64 * sizeof(_Float16));
    _Float16* W3h    = (_Float16*)alloc((size_t)128 * 64 * sizeof(_Float16));
    float*    stats  = (float*)alloc(512 * sizeof(float));  // S1,Q1,S2,Q2 (64ea) S3,Q3 (128ea)
    float*    scsh   = (float*)alloc(512 * sizeof(float));  // sc1,sh1,sc2,sh2 (64ea) sc3,sh3 (128ea)
    _Float16* regA   = (_Float16*)alloc((size_t)MM * KPAD1 * sizeof(_Float16)); // X0, later Y2
    _Float16* regB   = (_Float16*)alloc((size_t)MM * 128 * sizeof(_Float16));   // Y1, later Y3

    float* gS1 = stats;        float* gQ1 = stats + 64;
    float* gS2 = stats + 128;  float* gQ2 = stats + 192;
    float* gS3 = stats + 256;  float* gQ3 = stats + 384;
    float* sc1 = scsh;         float* sh1 = scsh + 64;
    float* sc2 = scsh + 128;   float* sh2 = scsh + 192;
    float* sc3 = scsh + 256;   float* sh3 = scsh + 384;

    _Float16* X0 = regA;
    _Float16* Y1 = regB;   // [M,64]
    _Float16* Y2 = regA;   // [M,64]  (X0 dead after gemm1)
    _Float16* Y3 = regB;   // [M,128] (Y1 dead after gemm2)

    const float invM = 1.0f / (float)MM;

    // 0) zero BN accumulators
    zero_stats_kernel<<<2, 256, 0, stream>>>(stats, 512);

    // 1) FPS -> new_xyz (d_out) + centers (ws)
    fps_kernel<<<BB, 256, 0, stream>>>(xyz, nxyz, out);

    // 2) ball query
    ballq_kernel<<<(BB * SS * 32 + 255) / 256, 256, 0, stream>>>(xyz, nxyz, idxbuf);

    // 3) f16 weights
    prep_weights_kernel<<<(64 * KPAD1 + 64 * 64 + 128 * 64 + 255) / 256, 256, 0, stream>>>(
        W1, W2, W3, W1h, W2h, W3h);

    // 4) gather/concat -> X0 [M,96] f16
    gather_kernel<<<(int)(((size_t)MM * 12 + 255) / 256), 256, 0, stream>>>(
        xyz, points, idxbuf, nxyz, X0);

    // 5) layer 1: [M,96] x [64,96]^T
    gemm_wmma_kernel<KPAD1, 64><<<MM / 128, 256, 0, stream>>>(X0, W1h, Y1, gS1, gQ1);
    bn_finalize_kernel<<<1, 64, 0, stream>>>(gS1, gQ1, g1, b1, sc1, sh1, 64, invM);
    bn_relu_kernel<<<(MM * 64 / 8 + 255) / 256, 256, 0, stream>>>(Y1, sc1, sh1, 64, MM * 64 / 8);

    // 6) layer 2: [M,64] x [64,64]^T
    gemm_wmma_kernel<64, 64><<<MM / 128, 256, 0, stream>>>(Y1, W2h, Y2, gS2, gQ2);
    bn_finalize_kernel<<<1, 64, 0, stream>>>(gS2, gQ2, g2, b2, sc2, sh2, 64, invM);
    bn_relu_kernel<<<(MM * 64 / 8 + 255) / 256, 256, 0, stream>>>(Y2, sc2, sh2, 64, MM * 64 / 8);

    // 7) layer 3: [M,64] x [128,64]^T
    gemm_wmma_kernel<64, 128><<<MM / 128, 256, 0, stream>>>(Y2, W3h, Y3, gS3, gQ3);
    bn_finalize_kernel<<<1, 128, 0, stream>>>(gS3, gQ3, g3, b3, sc3, sh3, 128, invM);

    // 8) fused BN+ReLU+maxpool -> new_points
    maxpool_kernel<<<(BB * SS * 128 + 255) / 256, 256, 0, stream>>>(Y3, sc3, sh3, out);

    (void)in_sizes; (void)n_in; (void)out_size; (void)ws_size;
}
